// TernaryFractalModel_80994493268390
// MI455X (gfx1250) — matrix operand order repaired and checked by reading, not compiled, not run on hardware
//
#include <hip/hip_runtime.h>
#include <hip/hip_bf16.h>
#include <stdint.h>

// ---------------------------------------------------------------------------
// Model dims (fixed by reference)
// ---------------------------------------------------------------------------
#define DMODEL 2048
#define NHEAD  16
#define HDIM   128
#define FFDIM  4096
#define VOCAB  32000
#define SEQ    1024
#define NSC    4
#define NIT    7
#define EPS_RMS 1.1920929e-07f

typedef __attribute__((ext_vector_type(8)))  __bf16       v8bf;
typedef __attribute__((ext_vector_type(16))) __bf16       v16bf;
typedef __attribute__((ext_vector_type(8)))  float        v8f;
typedef __attribute__((ext_vector_type(4)))  unsigned int v4u;
typedef __attribute__((ext_vector_type(8)))  int          v8i_t;
typedef __attribute__((ext_vector_type(4)))  int          v4i_t;

#if defined(__HIP_DEVICE_COMPILE__) && __has_builtin(__builtin_amdgcn_tensor_load_to_lds)
#define USE_TDM 1
#else
#define USE_TDM 0
#endif

// ---------------------------------------------------------------------------
// WMMA fragment helpers (bf16, 16x16x32, wave32)
//  A (16Mx32K): lanes 0-15 row M=lane,   K = {0..7, 16..23}
//               lanes 16-31 row M=lane-16, K = {8..15, 24..31}
//  B (32Kx16N): lanes 0-15 col N=lane,   K = {0..7, 16..23}
//               lanes 16-31 col N=lane-16, K = {8..15, 24..31}
// Both A rows and B columns are K-contiguous in our storage ([*,K] row-major).
// ---------------------------------------------------------------------------
__device__ __forceinline__ v16bf frag_a_row(const __bf16* row, int lane) {
  int k0 = (lane >> 4) * 8;
  v8bf lo = *(const v8bf*)(row + k0);
  v8bf hi = *(const v8bf*)(row + k0 + 16);
  union { v16bf v; v8bf h[2]; } u;
  u.h[0] = lo; u.h[1] = hi;
  return u.v;
}
__device__ __forceinline__ v16bf frag_b_col(const __bf16* col, int lane) {
  int k0 = (lane >> 4) * 8;
  v8bf lo = *(const v8bf*)(col + k0);
  v8bf hi = *(const v8bf*)(col + k0 + 16);
  union { v16bf v; v8bf h[2]; } u;
  u.h[0] = lo; u.h[1] = hi;
  return u.v;
}
__device__ __forceinline__ v8f wmma_bf16(v16bf a, v16bf b, v8f c) {
  return __builtin_amdgcn_wmma_f32_16x16x32_bf16(false, a, false, b, (short)0, c,
                                                 false, false);
}

// ---------------------------------------------------------------------------
// Tensor Data Mover: 2D tile (tile_rows x tile_k bf16 elems) global -> LDS
// D# layout per CDNA5 ISA 8.3/8.4 (group0 128b, group1 256b).
// This toolchain exposes the 6-arg builtin:
//   (uint32x4 g0, int32x8 g1, int32x4 g2, int32x4 g3, int32x8 g4, i32 cpol)
// Groups 2/3 zero => 2D tensor; trailing group zero-filled.
// ---------------------------------------------------------------------------
#if USE_TDM
__device__ __forceinline__ void tdm_load_tile_bf16(unsigned lds_off,
                                                   const __bf16* gsrc,
                                                   unsigned tile_k,
                                                   unsigned tile_rows,
                                                   unsigned row_stride_elems) {
  unsigned long long ga = (unsigned long long)(uintptr_t)gsrc;
  v4u g0;
  g0[0] = 1u;                                        // count=1, user descriptor
  g0[1] = lds_off;                                   // lds_addr (bytes)
  g0[2] = (unsigned)(ga & 0xffffffffu);              // global_addr[31:0]
  g0[3] = (unsigned)((ga >> 32) & 0x01ffffffu) | (2u << 30);  // addr[56:32]|type=2
  unsigned td0 = row_stride_elems;                   // tensor_dim0 (elems)
  unsigned td1 = 0x7fffffffu;                        // tensor_dim1 (large)
  unsigned long long st0 = (unsigned long long)row_stride_elems;
  v8i_t g1;
  g1[0] = (int)(1u << 16);                           // data_size=1 (2 bytes)
  g1[1] = (int)((td0 & 0xffffu) << 16);              // tensor_dim0[15:0]
  g1[2] = (int)(((td0 >> 16) & 0xffffu) | ((td1 & 0xffffu) << 16));
  g1[3] = (int)(((td1 >> 16) & 0xffffu) | ((tile_k & 0xffffu) << 16));
  g1[4] = (int)(tile_rows & 0xffffu);                // tile_dim1 (tile_dim2=0)
  g1[5] = (int)(st0 & 0xffffffffu);                  // tensor_dim0_stride[31:0]
  g1[6] = (int)((st0 >> 32) & 0xffffu);              // stride[47:32], dim1_stride=0
  g1[7] = 0;
  v4i_t z4; z4[0] = 0; z4[1] = 0; z4[2] = 0; z4[3] = 0;
  v8i_t z8;
#pragma unroll
  for (int i = 0; i < 8; ++i) z8[i] = 0;
  __builtin_amdgcn_tensor_load_to_lds(g0, g1, z4, z4, z8, 0);
}
#endif

// ---------------------------------------------------------------------------
// GEMM: out[M,N](f32) = epilogue( A[M,K](bf16) @ B[N,K](bf16)^T )
// modes: 0 plain, 1 *s[n], 2 res0 + *s[n], 3 res0 + is*(res1-res0 + *s[n])
// Block tile 128x128, 8 waves (2x4), wave tile 64x32, K-step 32, TDM staged.
// ---------------------------------------------------------------------------
__global__ __launch_bounds__(256) void k_gemm_bf16(
    const __bf16* __restrict__ A, const __bf16* __restrict__ B,
    const float* __restrict__ s, float* __restrict__ out,
    const float* __restrict__ res0, const float* __restrict__ res1,
    const float* __restrict__ iscp, int iscIdx, int M, int N, int K, int mode) {
  __shared__ alignas(16) __bf16 As[2][128 * 32];
  __shared__ alignas(16) __bf16 Bs[2][128 * 32];
  const int tid  = threadIdx.x;
  const int lane = tid & 31;
  const int wave = tid >> 5;
  const int bm = blockIdx.y, bn = blockIdx.x;
  const int wm = (wave >> 2) * 64;   // 0 or 64
  const int wn = (wave & 3) * 32;    // 0,32,64,96
  const size_t abase = (size_t)bm * 128 * K;
  const size_t bbase = (size_t)bn * 128 * K;

  v8f zero;
#pragma unroll
  for (int r = 0; r < 8; ++r) zero[r] = 0.0f;
  v8f acc[4][2];
#pragma unroll
  for (int i = 0; i < 4; ++i)
#pragma unroll
    for (int j = 0; j < 2; ++j) acc[i][j] = zero;

  const int KT = K / 32;

#if USE_TDM
  if (tid < 32) {  // wave 0 drives the Tensor Data Mover
    tdm_load_tile_bf16((unsigned)(uintptr_t)&As[0][0], A + abase, 32, 128, K);
    tdm_load_tile_bf16((unsigned)(uintptr_t)&Bs[0][0], B + bbase, 32, 128, K);
  }
#endif

  for (int kt = 0; kt < KT; ++kt) {
    const int cur = kt & 1;
#if USE_TDM
    if (tid < 32) {
      if (kt + 1 < KT) {
        const size_t ko = (size_t)(kt + 1) * 32;
        tdm_load_tile_bf16((unsigned)(uintptr_t)&As[cur ^ 1][0], A + abase + ko,
                           32, 128, K);
        tdm_load_tile_bf16((unsigned)(uintptr_t)&Bs[cur ^ 1][0], B + bbase + ko,
                           32, 128, K);
        __builtin_amdgcn_s_wait_tensorcnt(2);  // current pair complete
      } else {
        __builtin_amdgcn_s_wait_tensorcnt(0);
      }
    }
    __syncthreads();
#else
    for (int idx = tid; idx < 1024; idx += 256) {  // 128x32 bf16 as 8B chunks
      int r = idx >> 3, c4 = (idx & 7) * 4;
      *(uint64_t*)&As[0][r * 32 + c4] =
          *(const uint64_t*)&A[abase + (size_t)r * K + (size_t)kt * 32 + c4];
      *(uint64_t*)&Bs[0][r * 32 + c4] =
          *(const uint64_t*)&B[bbase + (size_t)r * K + (size_t)kt * 32 + c4];
    }
    __syncthreads();
#endif
    if (lane == 0 && kt + 2 < KT) {  // L2 prefetch two tiles ahead
      __builtin_prefetch(A + abase + (size_t)(kt + 2) * 32, 0, 0);
      __builtin_prefetch(B + bbase + (size_t)(kt + 2) * 32, 0, 0);
    }
    const int use = USE_TDM ? cur : 0;
    v16bf af[4], bf[2];
#pragma unroll
    for (int i = 0; i < 4; ++i)
      af[i] = frag_a_row(&As[use][(wm + i * 16 + (lane & 15)) * 32], lane);
#pragma unroll
    for (int j = 0; j < 2; ++j)
      bf[j] = frag_b_col(&Bs[use][(wn + j * 16 + (lane & 15)) * 32], lane);
#pragma unroll
    for (int i = 0; i < 4; ++i)
#pragma unroll
      for (int j = 0; j < 2; ++j) acc[i][j] = wmma_bf16(af[i], bf[j], acc[i][j]);
    __syncthreads();
  }

  // Epilogue. C/D layout: vgpr r, lanes 0-15 -> M=r, N=lane; lanes 16-31 -> M=r+8
  const int mrow0 = bm * 128 + wm + ((lane >> 4) * 8);
  const int ncol  = bn * 128 + wn + (lane & 15);
  const float is = (mode == 3) ? iscp[iscIdx] : 0.0f;
#pragma unroll
  for (int i = 0; i < 4; ++i) {
#pragma unroll
    for (int j = 0; j < 2; ++j) {
      const int n = ncol + j * 16;
      const float sc = (mode != 0) ? s[n] : 1.0f;
#pragma unroll
      for (int r = 0; r < 8; ++r) {
        const int m = mrow0 + i * 16 + r;
        const size_t o = (size_t)m * N + n;
        float v = acc[i][j][r] * sc;
        if (mode == 2) v += res0[o];
        else if (mode == 3) v = res0[o] + is * (res1[o] - res0[o] + v);
        out[o] = v;
      }
    }
  }
}

// ---------------------------------------------------------------------------
// Flash attention: block = (head, 128-query tile), 8 waves x 16 query rows.
// qkv f32 [SEQ, 3*DMODEL]; out bf16 [SEQ, DMODEL]. Causal, online softmax.
// ---------------------------------------------------------------------------
__global__ __launch_bounds__(256) void k_attn(const float* __restrict__ qkv,
                                              __bf16* __restrict__ attnb) {
  const int h  = blockIdx.x & (NHEAD - 1);
  const int qt = blockIdx.x >> 4;
  __shared__ alignas(16) __bf16 Qs[128][HDIM];
  __shared__ alignas(16) __bf16 Ks[32][HDIM];
  __shared__ alignas(16) __bf16 Vt[HDIM][32];
  __shared__ alignas(16) __bf16 Ps[8][16][32];
  const int tid = threadIdx.x, lane = tid & 31, wave = tid >> 5;
  const int DQ = 3 * DMODEL;
  const float qscale = 0.08838834764831845f;  // 1/sqrt(128)
  const float NEGBIG = -1.0e30f;

  for (int idx = tid; idx < 128 * HDIM; idx += 256) {
    int r = idx >> 7, c = idx & 127;
    Qs[r][c] = (__bf16)(qkv[(size_t)(qt * 128 + r) * DQ + h * HDIM + c] * qscale);
  }

  v8f o[8];
  float mrow[8], lrow[8];
#pragma unroll
  for (int n = 0; n < 8; ++n)
#pragma unroll
    for (int r = 0; r < 8; ++r) o[n][r] = 0.0f;
#pragma unroll
  for (int r = 0; r < 8; ++r) { mrow[r] = NEGBIG; lrow[r] = 0.0f; }

  const int qr0 = wave * 16;
  const int nkt = (qt + 1) * 4;  // key tiles of 32 up to causal frontier
  for (int kt = 0; kt < nkt; ++kt) {
    __syncthreads();
    const int kbase = kt * 32;
    for (int idx = tid; idx < 32 * HDIM; idx += 256) {
      int r = idx >> 7, c = idx & 127;
      const size_t rowb = (size_t)(kbase + r) * DQ + h * HDIM + c;
      Ks[r][c] = (__bf16)qkv[rowb + DMODEL];
      Vt[c][r] = (__bf16)qkv[rowb + 2 * DMODEL];
    }
    __syncthreads();

    v8f s0, s1;
#pragma unroll
    for (int r = 0; r < 8; ++r) { s0[r] = 0.0f; s1[r] = 0.0f; }
#pragma unroll
    for (int kc = 0; kc < 4; ++kc) {
      v16bf aq = frag_a_row(&Qs[qr0 + (lane & 15)][kc * 32], lane);
      v16bf b0 = frag_b_col(&Ks[0 + (lane & 15)][kc * 32], lane);
      v16bf b1 = frag_b_col(&Ks[16 + (lane & 15)][kc * 32], lane);
      s0 = wmma_bf16(aq, b0, s0);
      s1 = wmma_bf16(aq, b1, s1);
    }

    const int qrow0 = qt * 128 + qr0 + (lane >> 4) * 8;
    const int kcol  = kbase + (lane & 15);
#pragma unroll
    for (int r = 0; r < 8; ++r) {
      const int q = qrow0 + r;
      float e0 = (kcol <= q) ? s0[r] : NEGBIG;
      float e1 = (kcol + 16 <= q) ? s1[r] : NEGBIG;
      float tmax = fmaxf(e0, e1);
#pragma unroll
      for (int off = 8; off; off >>= 1)
        tmax = fmaxf(tmax, __shfl_xor(tmax, off, 16));
      const float mnew  = fmaxf(mrow[r], tmax);
      const float alpha = __expf(mrow[r] - mnew);
      const float p0 = __expf(e0 - mnew);
      const float p1 = __expf(e1 - mnew);
      float rs = p0 + p1;
#pragma unroll
      for (int off = 8; off; off >>= 1) rs += __shfl_xor(rs, off, 16);
      lrow[r] = lrow[r] * alpha + rs;
      mrow[r] = mnew;
#pragma unroll
      for (int n = 0; n < 8; ++n) o[n][r] *= alpha;
      const int pr = r + (lane >> 4) * 8;  // transpose P through LDS
      Ps[wave][pr][lane & 15] = (__bf16)p0;
      Ps[wave][pr][(lane & 15) + 16] = (__bf16)p1;
    }
    // per-wave LDS region; DS ops from same wave are in-order
    v16bf pf = frag_a_row(&Ps[wave][lane & 15][0], lane);
#pragma unroll
    for (int n = 0; n < 8; ++n) {
      v16bf vf = frag_b_col(&Vt[n * 16 + (lane & 15)][0], lane);
      o[n] = wmma_bf16(pf, vf, o[n]);
    }
  }

#pragma unroll
  for (int r = 0; r < 8; ++r) {
    const float inv = 1.0f / lrow[r];
    const int q = qt * 128 + qr0 + r + (lane >> 4) * 8;
#pragma unroll
    for (int n = 0; n < 8; ++n)
      attnb[(size_t)q * DMODEL + h * HDIM + n * 16 + (lane & 15)] =
          (__bf16)(o[n][r] * inv);
  }
}

// ---------------------------------------------------------------------------
// Elementwise / preprocessing kernels
// ---------------------------------------------------------------------------
__global__ void k_ternarize(const float* __restrict__ w, __bf16* __restrict__ o,
                            long long n) {
  long long i = (long long)blockIdx.x * blockDim.x + threadIdx.x;
  if (i < n) {
    float x = w[i];
    o[i] = (__bf16)((x > 0.5f) ? 1.0f : ((x < -0.5f) ? -1.0f : 0.0f));
  }
}
__global__ void k_cast_bf16(const float* __restrict__ w, __bf16* __restrict__ o,
                            long long n) {
  long long i = (long long)blockIdx.x * blockDim.x + threadIdx.x;
  if (i < n) o[i] = (__bf16)w[i];
}
__global__ void k_embed(const int* __restrict__ tok,
                        const float* __restrict__ emb, float* __restrict__ x) {
  const int t = blockIdx.x;
  const int tk = tok[t];
  for (int c = threadIdx.x; c < DMODEL; c += blockDim.x)
    x[(size_t)t * DMODEL + c] = emb[(size_t)tk * DMODEL + c];
}
__global__ __launch_bounds__(256) void k_rmsnorm(
    const float* __restrict__ x, const float* __restrict__ nw,
    const float* __restrict__ gamma, const float* __restrict__ beta,
    __bf16* __restrict__ out) {
  __shared__ float red[256];
  const int row = blockIdx.x, tid = threadIdx.x;
  const float* xr = x + (size_t)row * DMODEL;
  float ss = 0.0f;
  for (int c = tid; c < DMODEL; c += 256) { float v = xr[c]; ss += v * v; }
  red[tid] = ss;
  __syncthreads();
  for (int s2 = 128; s2; s2 >>= 1) {
    if (tid < s2) red[tid] += red[tid + s2];
    __syncthreads();
  }
  const float r = rsqrtf(red[0] / (float)DMODEL + EPS_RMS);
  for (int c = tid; c < DMODEL; c += 256) {
    float hv = xr[c] * r * nw[c];
    if (gamma) hv = hv * gamma[c] + beta[c];
    out[(size_t)row * DMODEL + c] = (__bf16)hv;
  }
}
__global__ void k_silu_mul(const float* __restrict__ g,
                           const float* __restrict__ u,
                           __bf16* __restrict__ o, long long n) {
  long long i = (long long)blockIdx.x * blockDim.x + threadIdx.x;
  if (i < n) {
    float gv = g[i];
    o[i] = (__bf16)((gv / (1.0f + __expf(-gv))) * u[i]);
  }
}

// ---------------------------------------------------------------------------
// Host orchestration
// ---------------------------------------------------------------------------
static inline size_t aup(size_t v) { return (v + 255) & ~(size_t)255; }

extern "C" void kernel_launch(void* const* d_in, const int* in_sizes, int n_in,
                              void* d_out, int out_size, void* d_ws,
                              size_t ws_size, hipStream_t stream) {
  const int*   tokens   = (const int*)d_in[0];
  const float* embed_w  = (const float*)d_in[1];
  const float* qkv_w    = (const float*)d_in[2];
  const float* qkv_s    = (const float*)d_in[3];
  const float* o_w      = (const float*)d_in[4];
  const float* o_s      = (const float*)d_in[5];
  const float* gate_w   = (const float*)d_in[6];
  const float* gate_s   = (const float*)d_in[7];
  const float* up_w     = (const float*)d_in[8];
  const float* up_s     = (const float*)d_in[9];
  const float* down_w   = (const float*)d_in[10];
  const float* down_s   = (const float*)d_in[11];
  const float* norm1_w  = (const float*)d_in[12];
  const float* norm2_w  = (const float*)d_in[13];
  const float* sgamma   = (const float*)d_in[14];
  const float* sbeta    = (const float*)d_in[15];
  const float* iter_sc  = (const float*)d_in[16];
  const float* norm_w   = (const float*)d_in[17];
  const float* lm_w     = (const float*)d_in[18];
  float* logits = (float*)d_out;
  (void)in_sizes; (void)n_in; (void)out_size; (void)ws_size;

  char* p = (char*)d_ws;
  size_t off = 0;
  auto take = [&](size_t bytes) { void* r = p + off; off = aup(off + bytes); return r; };
  __bf16* wq  = (__bf16*)take((size_t)3 * DMODEL * DMODEL * 2);
  __bf16* wo  = (__bf16*)take((size_t)DMODEL * DMODEL * 2);
  __bf16* wg  = (__bf16*)take((size_t)FFDIM * DMODEL * 2);
  __bf16* wu  = (__bf16*)take((size_t)FFDIM * DMODEL * 2);
  __bf16* wd  = (__bf16*)take((size_t)DMODEL * FFDIM * 2);
  __bf16* wlm = (__bf16*)take((size_t)VOCAB * DMODEL * 2);
  float*  xA   = (float*)take((size_t)SEQ * DMODEL * 4);
  float*  xB   = (float*)take((size_t)SEQ * DMODEL * 4);
  float*  x2   = (float*)take((size_t)SEQ * DMODEL * 4);
  __bf16* hb   = (__bf16*)take((size_t)SEQ * DMODEL * 2);
  float*  qkvb = (float*)take((size_t)SEQ * 3 * DMODEL * 4);
  __bf16* attb = (__bf16*)take((size_t)SEQ * DMODEL * 2);
  float*  gf   = (float*)take((size_t)SEQ * FFDIM * 4);
  float*  uf   = (float*)take((size_t)SEQ * FFDIM * 4);
  __bf16* actb = (__bf16*)take((size_t)SEQ * FFDIM * 2);

  auto ew = [&](long long n) { return dim3((unsigned)((n + 255) / 256)); };
  long long nq = (long long)3 * DMODEL * DMODEL, nd = (long long)DMODEL * DMODEL;
  long long nf = (long long)FFDIM * DMODEL, nl = (long long)VOCAB * DMODEL;
  k_ternarize<<<ew(nq), 256, 0, stream>>>(qkv_w, wq, nq);
  k_ternarize<<<ew(nd), 256, 0, stream>>>(o_w, wo, nd);
  k_ternarize<<<ew(nf), 256, 0, stream>>>(gate_w, wg, nf);
  k_ternarize<<<ew(nf), 256, 0, stream>>>(up_w, wu, nf);
  k_ternarize<<<ew(nf), 256, 0, stream>>>(down_w, wd, nf);
  k_cast_bf16<<<ew(nl), 256, 0, stream>>>(lm_w, wlm, nl);
  k_embed<<<SEQ, 256, 0, stream>>>(tokens, embed_w, xA);

  float* x = xA;
  float* xn = xB;
  for (int s = 0; s < NSC; ++s) {
    const float* gam = sgamma + (size_t)s * DMODEL;
    const float* bet = sbeta + (size_t)s * DMODEL;
    for (int it = 0; it < NIT; ++it) {
      k_rmsnorm<<<SEQ, 256, 0, stream>>>(x, norm1_w, gam, bet, hb);
      k_gemm_bf16<<<dim3(3 * DMODEL / 128, SEQ / 128), 256, 0, stream>>>(
          hb, wq, qkv_s, qkvb, nullptr, nullptr, nullptr, 0, SEQ, 3 * DMODEL,
          DMODEL, 1);
      k_attn<<<(SEQ / 128) * NHEAD, 256, 0, stream>>>(qkvb, attb);
      k_gemm_bf16<<<dim3(DMODEL / 128, SEQ / 128), 256, 0, stream>>>(
          attb, wo, o_s, x2, x, nullptr, nullptr, 0, SEQ, DMODEL, DMODEL, 2);
      k_rmsnorm<<<SEQ, 256, 0, stream>>>(x2, norm2_w, gam, bet, hb);
      k_gemm_bf16<<<dim3(FFDIM / 128, SEQ / 128), 256, 0, stream>>>(
          hb, wg, gate_s, gf, nullptr, nullptr, nullptr, 0, SEQ, FFDIM, DMODEL, 1);
      k_gemm_bf16<<<dim3(FFDIM / 128, SEQ / 128), 256, 0, stream>>>(
          hb, wu, up_s, uf, nullptr, nullptr, nullptr, 0, SEQ, FFDIM, DMODEL, 1);
      k_silu_mul<<<ew((long long)SEQ * FFDIM), 256, 0, stream>>>(
          gf, uf, actb, (long long)SEQ * FFDIM);
      // x_new = x + is * ((x2 - x) + down(act))
      k_gemm_bf16<<<dim3(DMODEL / 128, SEQ / 128), 256, 0, stream>>>(
          actb, wd, down_s, xn, x, x2, iter_sc, s * NIT + it, SEQ, DMODEL,
          FFDIM, 3);
      float* t = x; x = xn; xn = t;
    }
  }
  k_rmsnorm<<<SEQ, 256, 0, stream>>>(x, norm_w, nullptr, nullptr, hb);
  k_gemm_bf16<<<dim3(VOCAB / 128, SEQ / 128), 256, 0, stream>>>(
      hb, wlm, nullptr, logits, nullptr, nullptr, nullptr, 0, SEQ, VOCAB,
      DMODEL, 0);
}
